// LinearKAN_59167469470156
// MI455X (gfx1250) — matrix operand order repaired and checked by reading, not compiled, not run on hardware
//
#include <hip/hip_runtime.h>
#include <hip/hip_bf16.h>

typedef __attribute__((ext_vector_type(16))) _Float16 v16h;
typedef __attribute__((ext_vector_type(8)))  _Float16 v8h;
typedef __attribute__((ext_vector_type(8)))  float    v8f;

#define BATCH   64
#define IN_F    512
#define OUT_F   512
#define BASIS   8
#define KAUG    (IN_F * (BASIS + 1))   // 4608 augmented contraction axis
#define KSPLIT  16
#define KCHUNK  (KAUG / KSPLIT)        // 288 = 9 wmma k-steps of 32

#define A_BYTES ((size_t)BATCH * KAUG * 2)           // 589,824
#define B_BYTES ((size_t)OUT_F * KAUG * 2)           // 4,718,592
#define P_BYTES ((size_t)KSPLIT * BATCH * OUT_F * 4) // 2,097,152

// ---------------------------------------------------------------------------
// A[b, i*9 + r] : r=0 -> silu(x), r=1..8 -> T_r(clip(tanh(x)))   (f16)
// ---------------------------------------------------------------------------
__global__ void kan_prep_a(const float* __restrict__ x, _Float16* __restrict__ A) {
    int id = blockIdx.x * blockDim.x + threadIdx.x;
    if (id >= BATCH * IN_F) return;
    float xv = x[id];
    float silu = xv / (1.0f + __expf(-xv));
    float t = tanhf(xv);
    t = fminf(fmaxf(t, -1.0f + 1e-6f), 1.0f - 1e-6f);
    _Float16* row = A + (size_t)id * 9;   // id = b*512 + i, row-major over (b, i*9)
    row[0] = (_Float16)silu;
    float tkm1 = 1.0f, tk = t;
#pragma unroll
    for (int k = 1; k <= BASIS; ++k) {
        row[k] = (_Float16)tk;
        float tn = 2.0f * t * tk - tkm1;
        tkm1 = tk; tk = tn;
    }
}

// ---------------------------------------------------------------------------
// Bt[o, i*9 + r] : r=0 -> w[i,o], r=1..8 -> c[i,o,r-1]*w[i,o]    (f16, K-major
// per output column so every WMMA B-fragment load is one contiguous v16h)
// ---------------------------------------------------------------------------
__global__ void kan_prep_b(const float* __restrict__ w, const float* __restrict__ c,
                           _Float16* __restrict__ Bt) {
    int id = blockIdx.x * blockDim.x + threadIdx.x;
    if (id >= IN_F * OUT_F) return;
    int i = id >> 9, o = id & (OUT_F - 1);            // o fastest -> coalesced c reads
    float wv = w[(size_t)i * OUT_F + o];
    const float4* cp = (const float4*)(c + ((size_t)i * OUT_F + o) * BASIS);
    float4 c0 = cp[0], c1 = cp[1];
    _Float16* dst = Bt + (size_t)o * KAUG + (size_t)i * 9;
    dst[0] = (_Float16)wv;
    dst[1] = (_Float16)(c0.x * wv);
    dst[2] = (_Float16)(c0.y * wv);
    dst[3] = (_Float16)(c0.z * wv);
    dst[4] = (_Float16)(c0.w * wv);
    dst[5] = (_Float16)(c1.x * wv);
    dst[6] = (_Float16)(c1.y * wv);
    dst[7] = (_Float16)(c1.z * wv);
    dst[8] = (_Float16)(c1.w * wv);
}

// ---------------------------------------------------------------------------
// Split-K WMMA GEMM: one wave per block, full M=64 (4 fragments) x 16-wide
// N-tile, K chunk of 288 (9 steps of 32). 36 v_wmma per wave, explicitly
// double-buffered so every fragment load has a full k-step of latency cover.
// grid = (OUT_F/16, KSPLIT) = (32, 16) -> 512 waves.
// ---------------------------------------------------------------------------
__global__ __launch_bounds__(32) void kan_gemm(const _Float16* __restrict__ A,
                                               const _Float16* __restrict__ Bt,
                                               float* __restrict__ part) {
    const int o0   = blockIdx.x * 16;
    const int gy   = blockIdx.y;
    const int lane = threadIdx.x;
    const int n    = lane & 15;   // column / row-in-tile
    const int lh   = lane >> 4;   // which K half this lane owns
    const int kb   = gy * KCHUNK;

    v8f acc[4] = {};

    // B fragment (32x16 f16): lane holds col n, K = lh?16..31:0..15, contiguous.
    const _Float16* bp = Bt + (size_t)(o0 + n) * KAUG + kb + lh * 16;
    // A fragment (16x32 f16): lane holds row n; halves 0-15 = K(lh*16 ..),
    // contiguous (matches B): two adjacent 16B runs.
    const _Float16* ap = A + (size_t)n * KAUG + kb + lh * 16;

#define LOAD_B(dst, s) (dst) = *(const v16h*)(bp + (s) * 32)
#define LOAD_A(dst, s)                                                        \
    do {                                                                      \
        _Pragma("unroll")                                                     \
        for (int m_ = 0; m_ < 4; ++m_) {                                      \
            const _Float16* a0_ = ap + (size_t)m_ * 16 * KAUG + (s) * 32;     \
            v8h alo_ = *(const v8h*)a0_;                                      \
            v8h ahi_ = *(const v8h*)(a0_ + 8);                                \
            (dst)[m_] = __builtin_shufflevector(alo_, ahi_,                   \
                0, 1, 2, 3, 4, 5, 6, 7, 8, 9, 10, 11, 12, 13, 14, 15);        \
        }                                                                     \
    } while (0)

    v16h bcur, bnxt;
    v16h acur[4], anxt[4];
    LOAD_B(bcur, 0);
    LOAD_A(acur, 0);

#pragma unroll
    for (int s = 0; s < 9; ++s) {
        if (s + 1 < 9) {
            LOAD_B(bnxt, s + 1);
            LOAD_A(anxt, s + 1);
            __builtin_prefetch(bp + (s + 2) * 32, 0, 3);   // WGP-scope prefetch
        }
#pragma unroll
        for (int m = 0; m < 4; ++m)
            acc[m] = __builtin_amdgcn_wmma_f32_16x16x32_f16(
                false, acur[m], false, bcur, (short)0, acc[m], false, false);
        if (s + 1 < 9) {
            bcur = bnxt;
#pragma unroll
            for (int m = 0; m < 4; ++m) acur[m] = anxt[m];
        }
    }
#undef LOAD_A
#undef LOAD_B

    // D layout: VGPR v -> M = v + 8*lh, lane n -> N = n.
    float* P = part + (size_t)gy * BATCH * OUT_F + o0 + n;
#pragma unroll
    for (int m = 0; m < 4; ++m) {
#pragma unroll
        for (int v = 0; v < 8; ++v) {
            int brow = m * 16 + v + 8 * lh;
            P[(size_t)brow * OUT_F] = acc[m][v];
        }
    }
}

// ---------------------------------------------------------------------------
// Deterministic split-K reduction (no float atomics).
// ---------------------------------------------------------------------------
__global__ void kan_reduce(const float* __restrict__ part, float* __restrict__ out) {
    int id = blockIdx.x * blockDim.x + threadIdx.x;
    if (id >= BATCH * OUT_F) return;
    float s = 0.0f;
#pragma unroll
    for (int g = 0; g < KSPLIT; ++g)
        s += part[(size_t)g * BATCH * OUT_F + id];
    out[id] = s;
}

extern "C" void kernel_launch(void* const* d_in, const int* in_sizes, int n_in,
                              void* d_out, int out_size, void* d_ws, size_t ws_size,
                              hipStream_t stream) {
    const float* x = (const float*)d_in[0];   // (64, 512)
    const float* w = (const float*)d_in[1];   // (512, 512)
    const float* c = (const float*)d_in[2];   // (512, 512, 8)
    float* out = (float*)d_out;               // (64, 512) fp32

    char* ws = (char*)d_ws;
    _Float16* A    = (_Float16*)(ws);
    _Float16* Bt   = (_Float16*)(ws + A_BYTES);
    float*    part = (float*)(ws + A_BYTES + B_BYTES);

    kan_prep_a<<<(BATCH * IN_F + 255) / 256, 256, 0, stream>>>(x, A);
    kan_prep_b<<<(IN_F * OUT_F + 255) / 256, 256, 0, stream>>>(w, c, Bt);
    kan_gemm<<<dim3(OUT_F / 16, KSPLIT), 32, 0, stream>>>(A, Bt, part);
    kan_reduce<<<(BATCH * OUT_F + 255) / 256, 256, 0, stream>>>(part, out);
}
